// DecoderAttnRNN_32985348833419
// MI455X (gfx1250) — matrix operand order repaired and checked by reading, not compiled, not run on hardware
//
#include <hip/hip_runtime.h>
#include <math.h>

// Problem constants (match reference)
constexpr int B = 32, S = 64, T = 64, H = 512, E = 512, V = 32000;
constexpr int H3 = 3 * H;        // 1536
constexpr int XC = E + H;        // 1024

typedef __attribute__((ext_vector_type(16))) __bf16 v16bf;
typedef __attribute__((ext_vector_type(8)))  __bf16 v8bf;
typedef __attribute__((ext_vector_type(8)))  float  v8f;

__device__ __forceinline__ __bf16 f2bf(float f) {
    unsigned u = __float_as_uint(f);
    unsigned r = u + 0x7FFFu + ((u >> 16) & 1u);   // round-to-nearest-even
    unsigned short h = (unsigned short)(r >> 16);
    return __builtin_bit_cast(__bf16, h);
}

__device__ __forceinline__ float sigmoidf_(float x) { return 1.0f / (1.0f + expf(-x)); }

// ---------------------------------------------------------------------------
// Skinny GEMM: C[M,N] = A[M,K](bf16) * Bw[N,K]^T(bf16) + bias[N]
// Each wave owns a 16x64 strip of C (4 N-tiles sharing one A fragment).
// B weights are streamed Global -> LDS with GLOBAL_LOAD_ASYNC_TO_LDS_B128
// (ASYNCcnt ping-pong double buffering), then consumed as WMMA B fragments.
// Steady-state loop body is branch-free: fill(next) -> wait<=8 -> 4x WMMA.
// Requires M%16==0, N%64==0, K%32==0 (true for all call sites).
// ---------------------------------------------------------------------------
__global__ __launch_bounds__(128)
void wmma_gemm_bf16(const __bf16* __restrict__ Abf, int lda,
                    const __bf16* __restrict__ Bw,
                    const float* __restrict__ bias,
                    float* __restrict__ C, long ldc,
                    int M, int N, int K)
{
    // LDS: [buf][wave][4 tiles * 1KB]; tile = 16 cols x 64B (32 bf16 of K)
    __shared__ __align__(16) unsigned char ldsbuf[2][4][4096];

    const int lane = threadIdx.x & 31;
    const int wave = threadIdx.x >> 5;
    const int groupsN = N >> 6;                  // 16x64 strips per M-row
    const int unit = blockIdx.x * 4 + wave;      // wave-uniform
    if (unit >= (M >> 4) * groupsN) return;      // uniform exit: EXEC stays all-ones

    const int tm  = (unit / groupsN) << 4;
    const int tn0 = (unit % groupsN) << 6;

    const int half  = lane >> 4;                 // 0: lanes 0-15, 1: lanes 16-31
    const int col16 = lane & 15;
    const int rowA  = tm + col16;

    const __bf16* aRow = Abf + (size_t)rowA * lda;

    // async fill of one ping-pong buffer with the K-slab [kb, kb+32) for 4 tiles
    const int fcol   = lane >> 2;                // 0..7
    const int fchunk = lane & 3;                 // 16B chunk within a 64B col-slab
    auto fill = [&](int buf, int kb) {
        #pragma unroll
        for (int nt = 0; nt < 4; ++nt) {
            #pragma unroll
            for (int i = 0; i < 2; ++i) {
                int col = i * 8 + fcol;
                const __bf16* g = Bw + (size_t)(tn0 + nt * 16 + col) * K + kb + fchunk * 8;
                unsigned ldsoff =
                    (unsigned)(size_t)&ldsbuf[buf][wave][nt * 1024 + col * 64 + fchunk * 16];
                asm volatile("global_load_async_to_lds_b128 %0, %1, off"
                             :: "v"(ldsoff), "v"((unsigned long long)(size_t)g)
                             : "memory");
            }
        }
    };

    v8f acc0 = {}, acc1 = {}, acc2 = {}, acc3 = {};

    auto compute = [&](int cur, int ib) {
        // A fragment: 16x32 bf16, two contiguous 16B runs per lane
        const __bf16* ap = aRow + (ib << 5) + half * 16;
        v8bf alo = *(const v8bf*)ap;
        v8bf ahi = *(const v8bf*)(ap + 8);
        v16bf a = __builtin_shufflevector(alo, ahi, 0, 1, 2, 3, 4, 5, 6, 7,
                                          8, 9, 10, 11, 12, 13, 14, 15);

        const unsigned char* lb = &ldsbuf[cur][wave][col16 * 64 + half * 32];
        v16bf b0 = *(const v16bf*)(lb);
        v16bf b1 = *(const v16bf*)(lb + 1024);
        v16bf b2 = *(const v16bf*)(lb + 2048);
        v16bf b3 = *(const v16bf*)(lb + 3072);
        acc0 = __builtin_amdgcn_wmma_f32_16x16x32_bf16(false, a, false, b0, (short)0, acc0, false, false);
        acc1 = __builtin_amdgcn_wmma_f32_16x16x32_bf16(false, a, false, b1, (short)0, acc1, false, false);
        acc2 = __builtin_amdgcn_wmma_f32_16x16x32_bf16(false, a, false, b2, (short)0, acc2, false, false);
        acc3 = __builtin_amdgcn_wmma_f32_16x16x32_bf16(false, a, false, b3, (short)0, acc3, false, false);
    };

    const int nkb = K >> 5;                      // >= 1 for all call sites
    fill(0, 0);
    int ib = 0;
    for (; ib < nkb - 1; ++ib) {                 // steady state: no branches inside
        fill((ib + 1) & 1, (ib + 1) << 5);
        // the oldest 8 asyncs (current buffer) complete in order
        asm volatile("s_wait_asynccnt 0x8" ::: "memory");
        compute(ib & 1, ib);
    }
    asm volatile("s_wait_asynccnt 0x0" ::: "memory");
    compute(ib & 1, ib);

    // C/D layout: element r -> row tm + r + 8*half, col = tile_n + col16
    #pragma unroll
    for (int r = 0; r < 8; ++r) {
        int row = tm + r + 8 * half;
        float* cRow = C + (size_t)row * ldc + tn0 + col16;
        cRow[0]  = acc0[r] + bias[tn0 + col16];
        cRow[16] = acc1[r] + bias[tn0 + 16 + col16];
        cRow[32] = acc2[r] + bias[tn0 + 32 + col16];
        cRow[48] = acc3[r] + bias[tn0 + 48 + col16];
    }
}

// ---------------------------------------------------------------------------
// f32 -> bf16 conversion (grid-stride)
// ---------------------------------------------------------------------------
__global__ __launch_bounds__(256)
void f2bf_kernel(const float* __restrict__ src, __bf16* __restrict__ dst, int n)
{
    int stride = gridDim.x * blockDim.x;
    for (int i = blockIdx.x * blockDim.x + threadIdx.x; i < n; i += stride)
        dst[i] = f2bf(src[i]);
}

// ---------------------------------------------------------------------------
// init: h0 (f32 + bf16) = encoder_hidden[0], tok = sos
// ---------------------------------------------------------------------------
__global__ __launch_bounds__(256)
void init_kernel(const float* __restrict__ ehid, const int* __restrict__ sos,
                 float* __restrict__ h0, __bf16* __restrict__ h0bf,
                 int* __restrict__ tok)
{
    int i = blockIdx.x * blockDim.x + threadIdx.x;
    if (i < B * H) { float v = ehid[i]; h0[i] = v; h0bf[i] = f2bf(v); }
    if (i < B)     tok[i] = sos[0];
}

// ---------------------------------------------------------------------------
// x part of xc (bf16): xc_bf[b, 0:E] = bf16(emb[tok[b], :])
// ---------------------------------------------------------------------------
__global__ __launch_bounds__(256)
void embed_kernel(const float* __restrict__ emb, const int* __restrict__ tok,
                  __bf16* __restrict__ xc_bf)
{
    int b = blockIdx.x;
    int row = tok[b];
    for (int h = threadIdx.x; h < E; h += blockDim.x)
        xc_bf[(size_t)b * XC + h] = f2bf(emb[(size_t)row * E + h]);
}

// ---------------------------------------------------------------------------
// Bahdanau attention: scores = Va . tanh(q + keys_proj), softmax over S,
// context (bf16) -> xc_bf[b, E:E+H], weights -> attn_out[b, t, :]
// ---------------------------------------------------------------------------
__global__ __launch_bounds__(64)
void attn_kernel(const float* __restrict__ q, const float* __restrict__ kp,
                 const float* __restrict__ enc, const float* __restrict__ Va_w,
                 const float* __restrict__ Va_b,
                 __bf16* __restrict__ xc_bf, float* __restrict__ attn_out, int t)
{
    int b = blockIdx.x;
    int s = threadIdx.x;            // 0..S-1
    __shared__ float sw[S];

    const float* qb  = q  + (size_t)b * H;
    const float* kps = kp + ((size_t)b * S + s) * H;
    float sc = Va_b[0];
    for (int h = 0; h < H; ++h)
        sc += Va_w[h] * tanhf(qb[h] + kps[h]);
    sw[s] = sc;
    __syncthreads();

    float mx = -INFINITY;
    for (int i = 0; i < S; ++i) mx = fmaxf(mx, sw[i]);
    float sum = 0.0f;
    for (int i = 0; i < S; ++i) sum += expf(sw[i] - mx);
    float w = expf(sc - mx) / sum;
    __syncthreads();
    sw[s] = w;
    attn_out[((size_t)b * T + t) * S + s] = w;
    __syncthreads();

    float ctx[8] = {0, 0, 0, 0, 0, 0, 0, 0};
    for (int i = 0; i < S; ++i) {
        float wi = sw[i];
        const float* eb = enc + ((size_t)b * S + i) * H + s * 8;
        #pragma unroll
        for (int j = 0; j < 8; ++j) ctx[j] += wi * eb[j];
    }
    #pragma unroll
    for (int j = 0; j < 8; ++j)
        xc_bf[(size_t)b * XC + E + s * 8 + j] = f2bf(ctx[j]);
}

// ---------------------------------------------------------------------------
// GRU gates (r,z,n): h_new = (1-z)*n + z*h  (writes f32 + bf16)
// ---------------------------------------------------------------------------
__global__ __launch_bounds__(256)
void gru_kernel(const float* __restrict__ gi, const float* __restrict__ gh,
                const float* __restrict__ h,
                float* __restrict__ h_new, __bf16* __restrict__ h_new_bf)
{
    int idx = blockIdx.x * blockDim.x + threadIdx.x;
    if (idx >= B * H) return;
    int b = idx / H, j = idx % H;
    const float* gib = gi + (size_t)b * H3;
    const float* ghb = gh + (size_t)b * H3;
    float r = sigmoidf_(gib[j]         + ghb[j]);
    float z = sigmoidf_(gib[H + j]     + ghb[H + j]);
    float n = tanhf   (gib[2 * H + j] + r * ghb[2 * H + j]);
    float hv = (1.0f - z) * n + z * h[idx];
    h_new[idx] = hv;
    h_new_bf[idx] = f2bf(hv);
}

// ---------------------------------------------------------------------------
// Per-row argmax + in-place log_softmax over V; writes next token.
// ---------------------------------------------------------------------------
__global__ __launch_bounds__(256)
void finalize_kernel(float* __restrict__ row_base, long brow_stride,
                     int* __restrict__ tok)
{
    int b = blockIdx.x;
    float* row = row_base + (size_t)b * brow_stride;
    __shared__ float smax[256];
    __shared__ int   sidx[256];
    __shared__ float ssum[256];
    int tid = threadIdx.x;

    float best = -INFINITY; int bi = 0x7fffffff;
    for (int v = tid; v < V; v += 256) {
        float x = row[v];
        if (x > best || (x == best && v < bi)) { best = x; bi = v; }
    }
    smax[tid] = best; sidx[tid] = bi;
    __syncthreads();
    for (int s = 128; s > 0; s >>= 1) {
        if (tid < s) {
            float o = smax[tid + s]; int oi = sidx[tid + s];
            if (o > smax[tid] || (o == smax[tid] && oi < sidx[tid])) {
                smax[tid] = o; sidx[tid] = oi;
            }
        }
        __syncthreads();
    }
    float mx = smax[0];

    float acc = 0.0f;
    for (int v = tid; v < V; v += 256) acc += expf(row[v] - mx);
    ssum[tid] = acc;
    __syncthreads();
    for (int s = 128; s > 0; s >>= 1) {
        if (tid < s) ssum[tid] += ssum[tid + s];
        __syncthreads();
    }
    float lse = mx + logf(ssum[0]);

    for (int v = tid; v < V; v += 256) row[v] = row[v] - lse;
    if (tid == 0) tok[b] = sidx[0];
}

__global__ __launch_bounds__(256)
void copy_kernel(const float* __restrict__ src, float* __restrict__ dst, int n)
{
    int i = blockIdx.x * blockDim.x + threadIdx.x;
    if (i < n) dst[i] = src[i];
}

// ---------------------------------------------------------------------------
// Host launcher
// ---------------------------------------------------------------------------
static inline int cdiv_i(long a, long b) { return (int)((a + b - 1) / b); }

extern "C" void kernel_launch(void* const* d_in, const int* in_sizes, int n_in,
                              void* d_out, int out_size, void* d_ws, size_t ws_size,
                              hipStream_t stream) {
    const float* enc   = (const float*)d_in[0];   // [B,S,H]
    const float* ehid  = (const float*)d_in[1];   // [1,B,H]
    const float* emb   = (const float*)d_in[2];   // [V,E]
    const float* Wa_w  = (const float*)d_in[3];   // [H,H]
    const float* Wa_b  = (const float*)d_in[4];   // [H]
    const float* Ua_w  = (const float*)d_in[5];   // [H,H]
    const float* Ua_b  = (const float*)d_in[6];   // [H]
    const float* Va_w  = (const float*)d_in[7];   // [1,H]
    const float* Va_b  = (const float*)d_in[8];   // [1]
    const float* W_ih  = (const float*)d_in[9];   // [3H, E+H]
    const float* W_hh  = (const float*)d_in[10];  // [3H, H]
    const float* b_ih  = (const float*)d_in[11];  // [3H]
    const float* b_hh  = (const float*)d_in[12];  // [3H]
    const float* out_w = (const float*)d_in[13];  // [V,H]
    const float* out_b = (const float*)d_in[14];  // [V]
    const int*   sos   = (const int*)d_in[15];

    float* out       = (float*)d_out;
    float* out_logp  = out;                               // [B,T,V]
    float* out_h     = out + (size_t)B * T * V;           // [1,B,H]
    float* out_attn  = out_h + (size_t)B * H;             // [B,T,S]

    // Workspace carve-up (all regions 256B-aligned by construction)
    char* ws = (char*)d_ws;
    __bf16* outw_bf = (__bf16*)ws;  ws += (size_t)V * H * 2;        // 31.25 MB
    __bf16* Wih_bf  = (__bf16*)ws;  ws += (size_t)H3 * XC * 2;      //  3.0  MB
    __bf16* Whh_bf  = (__bf16*)ws;  ws += (size_t)H3 * H * 2;       //  1.5  MB
    __bf16* Wa_bf   = (__bf16*)ws;  ws += (size_t)H * H * 2;        //  0.5  MB
    __bf16* Ua_bf   = (__bf16*)ws;  ws += (size_t)H * H * 2;        //  0.5  MB
    __bf16* enc_bf  = (__bf16*)ws;  ws += (size_t)B * S * H * 2;    //  2.0  MB
    float*  kp      = (float*)ws;   ws += (size_t)B * S * H * 4;    //  4.0  MB
    float*  qbuf    = (float*)ws;   ws += (size_t)B * H * 4;
    __bf16* xc_bf   = (__bf16*)ws;  ws += (size_t)B * XC * 2;
    float*  gi      = (float*)ws;   ws += (size_t)B * H3 * 4;
    float*  gh      = (float*)ws;   ws += (size_t)B * H3 * 4;
    float*  hA      = (float*)ws;   ws += (size_t)B * H * 4;
    float*  hB      = (float*)ws;   ws += (size_t)B * H * 4;
    __bf16* hA_bf   = (__bf16*)ws;  ws += (size_t)B * H * 2;
    __bf16* hB_bf   = (__bf16*)ws;  ws += (size_t)B * H * 2;
    int*    tok     = (int*)ws;     ws += 256;

    auto gemm = [&](const __bf16* A, int lda, const __bf16* Bw, const float* bias,
                    float* C, long ldc, int M, int N, int K) {
        int units = (M >> 4) * (N >> 6);     // one 16x64 strip per wave
        int blocks = cdiv_i(units, 4);
        wmma_gemm_bf16<<<blocks, 128, 0, stream>>>(A, lda, Bw, bias, C, ldc, M, N, K);
    };

    // One-time: bf16 conversions + hoisted keys projection
    f2bf_kernel<<<2048, 256, 0, stream>>>(out_w, outw_bf, V * H);
    f2bf_kernel<<<1024, 256, 0, stream>>>(W_ih, Wih_bf, H3 * XC);
    f2bf_kernel<<<512, 256, 0, stream>>>(W_hh, Whh_bf, H3 * H);
    f2bf_kernel<<<256, 256, 0, stream>>>(Wa_w, Wa_bf, H * H);
    f2bf_kernel<<<256, 256, 0, stream>>>(Ua_w, Ua_bf, H * H);
    f2bf_kernel<<<1024, 256, 0, stream>>>(enc, enc_bf, B * S * H);
    // keys_proj[b*s, g] = enc[b*s, :] . Ua_w[g, :] + Ua_b[g]  (M=2048,N=512,K=512)
    gemm(enc_bf, H, Ua_bf, Ua_b, kp, H, B * S, H, H);

    init_kernel<<<cdiv_i(B * H, 256), 256, 0, stream>>>(ehid, sos, hA, hA_bf, tok);

    float*  h_cur = hA;     float*  h_nxt = hB;
    __bf16* hb_cur = hA_bf; __bf16* hb_nxt = hB_bf;
    for (int t = 0; t < T; ++t) {
        // q = h @ Wa^T + Wa_b                        [32,512]
        gemm(hb_cur, H, Wa_bf, Wa_b, qbuf, H, B, H, H);
        // x = emb[tok]  -> xc_bf[:, 0:E]
        embed_kernel<<<B, 256, 0, stream>>>(emb, tok, xc_bf);
        // attention: context -> xc_bf[:, E:], weights -> out_attn[:, t, :]
        attn_kernel<<<B, 64, 0, stream>>>(qbuf, kp, enc, Va_w, Va_b, xc_bf, out_attn, t);
        // gi = xc @ W_ih^T + b_ih                    [32,1536]
        gemm(xc_bf, XC, Wih_bf, b_ih, gi, H3, B, H3, XC);
        // gh = h @ W_hh^T + b_hh                     [32,1536]
        gemm(hb_cur, H, Whh_bf, b_hh, gh, H3, B, H3, H);
        // gates -> h_nxt (+bf16)
        gru_kernel<<<cdiv_i(B * H, 256), 256, 0, stream>>>(gi, gh, h_cur, h_nxt, hb_nxt);
        // logits = h_nxt @ out_w^T + out_b -> out_logp[:, t, :]
        gemm(hb_nxt, H, outw_bf, out_b, out_logp + (size_t)t * V, (long)T * V, B, V, H);
        // argmax -> tok, in-place log_softmax
        finalize_kernel<<<B, 256, 0, stream>>>(out_logp + (size_t)t * V, (long)T * V, tok);
        // swap h buffers
        float*  tf = h_cur;  h_cur = h_nxt;   h_nxt = tf;
        __bf16* tb = hb_cur; hb_cur = hb_nxt; hb_nxt = tb;
    }

    copy_kernel<<<cdiv_i(B * H, 256), 256, 0, stream>>>(h_cur, out_h, B * H);
}
